// Node_Aggregator_70463233458807
// MI455X (gfx1250) — matrix operand order repaired and checked by reading, not compiled, not run on hardware
//
#include <hip/hip_runtime.h>
#include <hip/hip_bf16.h>
#include <math.h>

typedef __attribute__((ext_vector_type(16))) _Float16 v16h;
typedef __attribute__((ext_vector_type(2)))  _Float16 v2h;
typedef __attribute__((ext_vector_type(8)))  float    v8f;

#define HH 50
#define AA 16
#define LL 66
#define DD 64

// ---------------------------------------------------------------------------
// Kernel 1: Q[N,64] = cat(self,target)[N,128] @ W[128,64] + b   via WMMA f16
//
// W is staged to LDS pre-permuted into WMMA B-fragment order so each lane
// reads its whole 16-half fragment as two ds_load_b128.
//
// Fragment-space index: i = ((kc*4 + t)*32 + lane)*16 + h
//   maps to W[(kc*32 + (h&7) + 8*(lane>>4) + 16*(h>>3))*64 + t*16 + (lane&15)]
//
// A fragment layout (16-bit A 16x32): lane holds row M = lane&15;
//   lane<16:  halfs 0..7 -> K=0..7,   halfs 8..15 -> K=16..23
//   lane>=16: halfs 0..7 -> K=8..15,  halfs 8..15 -> K=24..31
// C/D layout: VGPR r -> M = r + 8*(lane>>4), N = lane&15
// ---------------------------------------------------------------------------
__global__ void __launch_bounds__(256)
query_gemm_wmma(const float* __restrict__ self_f,
                const float* __restrict__ tgt_f,
                const float* __restrict__ W,     // [128,64]
                const float* __restrict__ bias,  // [64]
                float* __restrict__ Q,           // [N,64]
                int N)
{
    __shared__ __align__(32) _Float16 sWf[128 * 64];   // 16 KB, fragment order

    const int tid = threadIdx.x;

    // ---- stage W -> LDS in B-fragment order, packed 2 halfs per store
    // pair index p in [0, 4096): p = ((kc*4+t)*32 + ln)*8 + hp,  h = 2*hp
    for (int p = tid; p < 4096; p += 256) {
        const int hp  = p & 7;
        const int ln  = (p >> 3) & 31;
        const int tt  = (p >> 8) & 3;
        const int kc  = p >> 10;
        const int col = tt * 16 + (ln & 15);
        const int kk  = kc * 32 + 2 * (hp & 3) + 8 * (ln >> 4) + 16 * (hp >> 2);
        v2h pk;
        pk[0] = (_Float16)W[(size_t)kk * 64 + col];
        pk[1] = (_Float16)W[(size_t)(kk + 1) * 64 + col];
        ((v2h*)sWf)[p] = pk;
    }
    __syncthreads();

    const int wave = tid >> 5;
    const int lane = tid & 31;
    const int m    = lane & 15;
    const int hi   = lane >> 4;              // 0 or 1
    const int rowBase = (blockIdx.x * 8 + wave) * 16;
    if (rowBase >= N) return;

    const int row = rowBase + m;

    v8f acc0 = {}, acc1 = {}, acc2 = {}, acc3 = {};

    #pragma unroll
    for (int kc = 0; kc < 4; ++kc) {
        const int kbase = kc * 32;
        // ---- A fragment: X[row][kbase .. kbase+31] (X = cat(self,target))
        const float* src = (kbase < 64) ? (self_f + (size_t)row * 64 + kbase)
                                        : (tgt_f  + (size_t)row * 64 + (kbase - 64));
        const int off0 = hi * 8;
        v16h a;
        #pragma unroll
        for (int h = 0; h < 8; ++h) a[h]     = (_Float16)src[off0 + h];
        #pragma unroll
        for (int h = 0; h < 8; ++h) a[h + 8] = (_Float16)src[off0 + 16 + h];

        // ---- B fragments: contiguous 32 B per lane -> 2x ds_load_b128 each
        const v16h b0 = *(const v16h*)&sWf[((kc * 4 + 0) * 32 + lane) * 16];
        const v16h b1 = *(const v16h*)&sWf[((kc * 4 + 1) * 32 + lane) * 16];
        const v16h b2 = *(const v16h*)&sWf[((kc * 4 + 2) * 32 + lane) * 16];
        const v16h b3 = *(const v16h*)&sWf[((kc * 4 + 3) * 32 + lane) * 16];

        acc0 = __builtin_amdgcn_wmma_f32_16x16x32_f16(false, a, false, b0, (short)0, acc0, false, false);
        acc1 = __builtin_amdgcn_wmma_f32_16x16x32_f16(false, a, false, b1, (short)0, acc1, false, false);
        acc2 = __builtin_amdgcn_wmma_f32_16x16x32_f16(false, a, false, b2, (short)0, acc2, false, false);
        acc3 = __builtin_amdgcn_wmma_f32_16x16x32_f16(false, a, false, b3, (short)0, acc3, false, false);
    }

    // ---- bias + store: D element (r, lane) = C[M = r + 8*hi][N = m]
    #pragma unroll
    for (int r = 0; r < 8; ++r) {
        const int orow = rowBase + r + hi * 8;
        float* qp = Q + (size_t)orow * 64 + m;
        qp[0]  = acc0[r] + bias[m];
        qp[16] = acc1[r] + bias[m + 16];
        qp[32] = acc2[r] + bias[m + 32];
        qp[48] = acc3[r] + bias[m + 48];
    }
}

// ---------------------------------------------------------------------------
// Deterministic Gumbel noise from an integer hash (threefry not reproducible)
// ---------------------------------------------------------------------------
__device__ __forceinline__ float gumbel_noise(unsigned idx)
{
    unsigned x = idx * 0x9E3779B9u + 0x7F4A7C15u;
    x ^= x >> 16; x *= 0x85EBCA6Bu;
    x ^= x >> 13; x *= 0xC2B2AE35u;
    x ^= x >> 16;
    float u = (float)(x | 1u) * 2.3283064365386963e-10f;   // (0,1)
    return -logf(-logf(u));
}

// ---------------------------------------------------------------------------
// Kernel 2: per-node gather + Gumbel-top-k select + attention aggregate.
// One 256-thread (8-wave) workgroup per node. Single pass over the 66
// neighbor rows computes distance + both attention dots and stages rows in
// LDS for the output reduction (selection set is order-invariant downstream).
// ---------------------------------------------------------------------------
__global__ void __launch_bounds__(256)
node_aggregate(const float* __restrict__ Q,        // [N,64]
               const int*   __restrict__ hist_uv,  // [N,50]
               const int*   __restrict__ hist_r,   // [N,50]
               const int*   __restrict__ adj,      // [N,16]
               const float* __restrict__ u2e,      // [VU,64]
               const float* __restrict__ v2e,      // [VV,64]
               const float* __restrict__ r2e,      // [8,64]
               const float* __restrict__ ratt,     // [128]
               const int*   __restrict__ k_sel,    // [1]
               float* __restrict__ out)            // [N,64]
{
    __shared__ float sE[LL][DD];     // gathered neighbor rows (16.9 KB)
    __shared__ float sQ[DD];
    __shared__ float sRatt[2 * DD];
    __shared__ float sScore[LL];
    __shared__ float sAtt[LL];
    __shared__ float sAttM[LL];
    __shared__ float sWgt[LL];

    const int n    = blockIdx.x;
    const int tid  = threadIdx.x;
    const int lane = tid & 31;
    const int wave = tid >> 5;

    if (tid < DD) sQ[tid] = Q[(size_t)n * DD + tid];
    if (tid < 2 * DD) sRatt[tid] = ratt[tid];
    __syncthreads();

    const float q0 = sQ[2 * lane], q1 = sQ[2 * lane + 1];
    const float ra0 = sRatt[2 * lane],      ra1 = sRatt[2 * lane + 1];
    const float rb0 = sRatt[DD + 2 * lane], rb1 = sRatt[DD + 2 * lane + 1];

    // ---- pass 1: gather rows, distances, attention dots (wave per l)
    for (int l = wave; l < LL; l += 8) {
        // prefetch next iteration's embedding row
        const int ln = l + 8;
        if (ln < LL) {
            const float* nsrc = (ln < HH)
                ? u2e + (size_t)hist_uv[n * HH + ln] * DD
                : v2e + (size_t)adj[n * AA + (ln - HH)] * DD;
            __builtin_prefetch(nsrc + 2 * lane, 0, 3);
        }

        const float* src;
        const float* rr;
        if (l < HH) {
            src = u2e + (size_t)hist_uv[n * HH + l] * DD;
            rr  = r2e + (size_t)hist_r[n * HH + l] * DD;
        } else {
            src = v2e + (size_t)adj[n * AA + (l - HH)] * DD;
            rr  = r2e + (size_t)7 * DD;                 // REL_TOK
        }

        const float e0 = src[2 * lane], e1 = src[2 * lane + 1];
        const float r0 = rr[2 * lane],  r1 = rr[2 * lane + 1];
        sE[l][2 * lane]     = e0;
        sE[l][2 * lane + 1] = e1;

        float d2 = (q0 - e0) * (q0 - e0) + (q1 - e1) * (q1 - e1);
        float ap = e0 * ra0 + e1 * ra1 + r0 * rb0 + r1 * rb1;
        #pragma unroll
        for (int off = 16; off > 0; off >>= 1) {
            d2 += __shfl_xor(d2, off, 32);
            ap += __shfl_xor(ap, off, 32);
        }
        if (lane == 0) {
            sScore[l] = -sqrtf(d2) + gumbel_noise((unsigned)(n * LL + l));
            sAtt[l]   = ap;
        }
    }
    __syncthreads();

    // ---- selection by rank (Gumbel-top-k; set membership only)
    const int k = *k_sel;
    if (tid < LL) {
        const float s = sScore[tid];
        int rank = 0;
        for (int j = 0; j < LL; ++j) {
            const float sj = sScore[j];
            rank += (sj > s) || (sj == s && j < tid);
        }
        sAttM[tid] = (rank < k) ? sAtt[tid] : -INFINITY;
    }
    __syncthreads();

    // ---- softmax over selected (redundant per-thread scan of 66 entries)
    if (tid < LL) {
        float amax = -INFINITY;
        for (int j = 0; j < LL; ++j) amax = fmaxf(amax, sAttM[j]);
        float sum = 0.0f;
        for (int j = 0; j < LL; ++j) sum += expf(sAttM[j] - amax);
        sWgt[tid] = expf(sAttM[tid] - amax) / sum;     // 0 for unselected
    }
    __syncthreads();

    // ---- weighted sum: out[n][d] = sum_l w[l] * e[l][d]
    if (tid < DD) {
        float acc = 0.0f;
        #pragma unroll 6
        for (int l = 0; l < LL; ++l)
            acc = fmaf(sWgt[l], sE[l][tid], acc);
        out[(size_t)n * DD + tid] = acc;
    }
}

// ---------------------------------------------------------------------------
extern "C" void kernel_launch(void* const* d_in, const int* in_sizes, int n_in,
                              void* d_out, int out_size, void* d_ws, size_t ws_size,
                              hipStream_t stream)
{
    const float* self_f  = (const float*)d_in[0];
    const float* tgt_f   = (const float*)d_in[1];
    const int*   hist_uv = (const int*)  d_in[2];
    const int*   hist_r  = (const int*)  d_in[3];
    const int*   adj     = (const int*)  d_in[4];
    const float* u2e     = (const float*)d_in[5];
    const float* v2e     = (const float*)d_in[6];
    const float* r2e     = (const float*)d_in[7];
    const float* ratt    = (const float*)d_in[8];
    const float* W       = (const float*)d_in[9];
    const float* bias    = (const float*)d_in[10];
    const int*   k_sel   = (const int*)  d_in[11];

    const int N = in_sizes[0] / DD;

    float* Q = (float*)d_ws;     // N*64 floats of scratch

    query_gemm_wmma<<<dim3((N + 127) / 128), 256, 0, stream>>>(
        self_f, tgt_f, W, bias, Q, N);

    node_aggregate<<<dim3(N), 256, 0, stream>>>(
        Q, hist_uv, hist_r, adj, u2e, v2e, r2e, ratt, k_sel, (float*)d_out);
}
